// V_Std_52123723105097
// MI455X (gfx1250) — matrix operand order repaired and checked by reading, not compiled, not run on hardware
//
#include <hip/hip_runtime.h>
#include <hip/hip_bf16.h>
#include <math.h>

typedef __attribute__((ext_vector_type(16))) _Float16 v16h;
typedef __attribute__((ext_vector_type(8)))  _Float16 v8h;
typedef __attribute__((ext_vector_type(8)))  float    v8f;

#define DD 64
#define NEG_SLOPE 0.2f
#define EPSV 1e-5f

// ---------------- WMMA helpers (gfx1250, wave32) ----------------

__device__ __forceinline__ v8f wmma_f16(v16h a, v16h b, v8f c) {
  // v_wmma_f32_16x16x32_f16: D = A(16x32) x B(32x16) + C(16x16 f32)
  return __builtin_amdgcn_wmma_f32_16x16x32_f16(false, a, false, b, (short)0, c,
                                                false, false);
}

// A fragment (16x32 f16) from row-major fp32 [rows][64], fused leaky-relu.
// ISA layout: lanes 0-15 hold row M=lane, halves 0..7 = K k0..k0+7,
// halves 8..15 = K k0+16..k0+23; lanes 16-31 same rows, K shifted by +8.
__device__ __forceinline__ v16h load_a_global_f32(const float* __restrict__ base,
                                                  int m_base, int rows, int k0,
                                                  int lane, float slope) {
  int m = m_base + (lane & 15);
  if (m >= rows) m = rows - 1;            // clamp; OOB rows predicated at store
  const float* p = base + (size_t)m * DD;
  int ka = k0 + (lane >> 4) * 8;
  v16h out;
#pragma unroll
  for (int i = 0; i < 8; ++i) {
    float x = p[ka + i];
    x = (x >= 0.f) ? x : slope * x;
    out[i] = (_Float16)x;
    float y = p[ka + 16 + i];
    y = (y >= 0.f) ? y : slope * y;
    out[8 + i] = (_Float16)y;
  }
  return out;
}

// A fragment from LDS f16 tile [16][64] (row-major).
__device__ __forceinline__ v16h load_a_lds(const _Float16* lds, int k0, int lane) {
  int m = lane & 15;
  int ka = k0 + (lane >> 4) * 8;
  const _Float16* p = lds + m * DD + ka;
  v8h lo = *(const v8h*)p;          // K ka..ka+7   (16B aligned)
  v8h hi = *(const v8h*)(p + 16);   // K ka+16..+23
  v16h out;
#pragma unroll
  for (int i = 0; i < 8; ++i) { out[i] = lo[i]; out[8 + i] = hi[i]; }
  return out;
}

// B fragment (32x16 f16) for x @ W^T: B[k][n] = W[n][k], W row-major f16 [64][64].
// ISA layout: lanes 0-15 hold N=lane, halves 0..7 = K k0..k0+7, halves 8..15 = K k0+16..k0+23;
// lanes 16-31 same, K shifted by +8.
__device__ __forceinline__ v16h load_b_w16(const _Float16* __restrict__ w16,
                                           int n0, int k0, int lane) {
  int n = n0 + (lane & 15);
  int kb = k0 + (lane >> 4) * 8;
  const _Float16* p = w16 + n * DD + kb;
  v8h lo = *(const v8h*)p;          // 16B aligned
  v8h hi = *(const v8h*)(p + 16);
  v16h out;
#pragma unroll
  for (int i = 0; i < 8; ++i) { out[i] = lo[i]; out[8 + i] = hi[i]; }
  return out;
}

// Store C/D fragment (+bias, leaky-relu with given slope; slope=0 => relu)
// into LDS f16 tile [16][64]. C layout: elem r of lane L -> row 8*(L>>4)+r, col L&15.
__device__ __forceinline__ void store_c_lds_act(_Float16* lds, int n0, v8f c,
                                                const float* __restrict__ bias,
                                                int lane, float slope) {
  int n = n0 + (lane & 15);
  float b = bias[n];
  int mb = (lane >> 4) * 8;
#pragma unroll
  for (int r = 0; r < 8; ++r) {
    float x = c[r] + b;
    x = (x >= 0.f) ? x : slope * x;
    lds[(mb + r) * DD + n] = (_Float16)x;
  }
}

// ---------------- kernels ----------------

// Convert the six 64x64 fp32 weight matrices to f16 (L2-resident, reused by all waves).
__global__ void convert_w_kernel(const float* __restrict__ a, const float* __restrict__ b,
                                 const float* __restrict__ c, const float* __restrict__ d,
                                 const float* __restrict__ e, const float* __restrict__ f,
                                 _Float16* __restrict__ out) {
  int i = blockIdx.x * blockDim.x + threadIdx.x;
  if (i < 4096) {
    out[0 * 4096 + i] = (_Float16)a[i];
    out[1 * 4096 + i] = (_Float16)b[i];
    out[2 * 4096 + i] = (_Float16)c[i];
    out[3 * 4096 + i] = (_Float16)d[i];
    out[4 * 4096 + i] = (_Float16)e[i];
    out[5 * 4096 + i] = (_Float16)f[i];
  }
}

__global__ void zero_kernel(float* __restrict__ p, size_t n) {
  size_t i = (size_t)blockIdx.x * blockDim.x + threadIdx.x;
  size_t stride = (size_t)gridDim.x * blockDim.x;
  for (; i < n; i += stride) p[i] = 0.f;
}

// Vp = lrelu(lrelu(V) @ pA_w^T + pA_b) @ pB_w^T + pB_b   (one wave32 / 16-node tile)
__global__ void __launch_bounds__(32)
node_pool_kernel(const float* __restrict__ V, const _Float16* __restrict__ w16,
                 const float* __restrict__ pA_b, const float* __restrict__ pB_b,
                 float* __restrict__ Vp, int N) {
  __shared__ __align__(32) _Float16 lds[16 * DD];
  int lane = threadIdx.x & 31;
  int m_base = blockIdx.x * 16;
  if (m_base >= N) return;
  const _Float16* wA = w16 + 0 * 4096;
  const _Float16* wB = w16 + 1 * 4096;

  v8f acc[4] = {};
#pragma unroll
  for (int kk = 0; kk < 2; ++kk) {
    v16h a = load_a_global_f32(V, m_base, N, kk * 32, lane, NEG_SLOPE); // act(V)
#pragma unroll
    for (int nt = 0; nt < 4; ++nt)
      acc[nt] = wmma_f16(a, load_b_w16(wA, nt * 16, kk * 32, lane), acc[nt]);
  }
#pragma unroll
  for (int nt = 0; nt < 4; ++nt)
    store_c_lds_act(lds, nt * 16, acc[nt], pA_b, lane, NEG_SLOPE);
  __syncthreads();

  v8f acc2[4] = {};
#pragma unroll
  for (int kk = 0; kk < 2; ++kk) {
    v16h a = load_a_lds(lds, kk * 32, lane);
#pragma unroll
    for (int nt = 0; nt < 4; ++nt)
      acc2[nt] = wmma_f16(a, load_b_w16(wB, nt * 16, kk * 32, lane), acc2[nt]);
  }
  int mb = m_base + (lane >> 4) * 8;
  int nlo = lane & 15;
#pragma unroll
  for (int nt = 0; nt < 4; ++nt) {
    int n = nt * 16 + nlo;
    float b = pB_b[n];
#pragma unroll
    for (int r = 0; r < 8; ++r) {
      int m = mb + r;
      if (m < N) Vp[(size_t)m * DD + n] = acc2[nt][r] + b;
    }
  }
}

__global__ void degree_kernel(const int* __restrict__ dst, float* __restrict__ deg,
                              int NE) {
  int i = blockIdx.x * blockDim.x + threadIdx.x;
  if (i < NE) atomicAdd(deg + dst[i], 1.0f);
}

// Fully fused edge pipeline: E -> MLP(2x) -> (sigmoid scale, shift) -> gather Vp[src]
// -> M1, M1^2 -> atomic scatter onto dst sums. One wave32 per 16-edge tile.
__global__ void __launch_bounds__(32)
edge_msg_kernel(const float* __restrict__ E, const int* __restrict__ src,
                const int* __restrict__ dst, const _Float16* __restrict__ w16,
                const float* __restrict__ mb1, const float* __restrict__ mb2,
                const float* __restrict__ Bb, const float* __restrict__ Cb,
                const float* __restrict__ Vp,
                float* __restrict__ V1s, float* __restrict__ V2s, int NE) {
  __shared__ __align__(32) _Float16 lds[16 * DD];
  int lane = threadIdx.x & 31;
  int e0 = blockIdx.x * 16;
  if (e0 >= NE) return;
  const _Float16* w1 = w16 + 2 * 4096;
  const _Float16* w2 = w16 + 3 * 4096;
  const _Float16* wS = w16 + 4 * 4096;
  const _Float16* wC = w16 + 5 * 4096;

  // h1 = relu(E @ W1^T + b1)
  v8f acc[4] = {};
#pragma unroll
  for (int kk = 0; kk < 2; ++kk) {
    v16h a = load_a_global_f32(E, e0, NE, kk * 32, lane, 1.0f); // identity
#pragma unroll
    for (int nt = 0; nt < 4; ++nt)
      acc[nt] = wmma_f16(a, load_b_w16(w1, nt * 16, kk * 32, lane), acc[nt]);
  }
#pragma unroll
  for (int nt = 0; nt < 4; ++nt)
    store_c_lds_act(lds, nt * 16, acc[nt], mb1, lane, 0.0f);   // relu
  __syncthreads();

  // h2 = relu(h1 @ W2^T + b2)
  v8f acc2[4] = {};
#pragma unroll
  for (int kk = 0; kk < 2; ++kk) {
    v16h a = load_a_lds(lds, kk * 32, lane);
#pragma unroll
    for (int nt = 0; nt < 4; ++nt)
      acc2[nt] = wmma_f16(a, load_b_w16(w2, nt * 16, kk * 32, lane), acc2[nt]);
  }
  __syncthreads();
#pragma unroll
  for (int nt = 0; nt < 4; ++nt)
    store_c_lds_act(lds, nt * 16, acc2[nt], mb2, lane, 0.0f);  // relu
  __syncthreads();

  // scale-logits = h2 @ BW^T ; shift = h2 @ CW^T (biases applied below)
  v8f sA[4] = {}, tA[4] = {};
#pragma unroll
  for (int kk = 0; kk < 2; ++kk) {
    v16h a = load_a_lds(lds, kk * 32, lane);
#pragma unroll
    for (int nt = 0; nt < 4; ++nt) {
      sA[nt] = wmma_f16(a, load_b_w16(wS, nt * 16, kk * 32, lane), sA[nt]);
      tA[nt] = wmma_f16(a, load_b_w16(wC, nt * 16, kk * 32, lane), tA[nt]);
    }
  }

  int nlo = lane & 15;
  float bS[4], bC[4];
#pragma unroll
  for (int nt = 0; nt < 4; ++nt) { bS[nt] = Bb[nt * 16 + nlo]; bC[nt] = Cb[nt * 16 + nlo]; }

  int eb = e0 + (lane >> 4) * 8;
#pragma unroll
  for (int r = 0; r < 8; ++r) {
    int e = eb + r;
    if (e < NE) {
      int s = src[e], d2 = dst[e];
      const float* vp = Vp + (size_t)s * DD;     // gather (L2-resident)
      float* p1 = V1s + (size_t)d2 * DD;
      float* p2 = V2s + (size_t)d2 * DD;
#pragma unroll
      for (int nt = 0; nt < 4; ++nt) {
        int n = nt * 16 + nlo;
        float sg = 1.f / (1.f + __expf(-(sA[nt][r] + bS[nt])));
        float m1 = sg * vp[n] + (tA[nt][r] + bC[nt]);
        atomicAdd(p1 + n, m1);                   // global_atomic_add_f32 -> L2
        atomicAdd(p2 + n, m1 * m1);
      }
    }
  }
}

// out = sqrt(relu((V2sum - V1sum)/max(deg,1)) + eps)
__global__ void finalize_kernel(const float* __restrict__ V1s,
                                const float* __restrict__ V2s,
                                const float* __restrict__ deg,
                                float* __restrict__ out, int N) {
  int i = blockIdx.x * blockDim.x + threadIdx.x;
  if (i < N * DD) {
    int node = i >> 6;
    float denom = fmaxf(deg[node], 1.f);
    float v = (V2s[i] - V1s[i]) / denom;
    out[i] = sqrtf(fmaxf(v, 0.f) + EPSV);
  }
}

// ---------------- launch ----------------

extern "C" void kernel_launch(void* const* d_in, const int* in_sizes, int n_in,
                              void* d_out, int out_size, void* d_ws, size_t ws_size,
                              hipStream_t stream) {
  const float* V    = (const float*)d_in[0];
  const float* E    = (const float*)d_in[1];
  const int*   src  = (const int*)d_in[2];
  const int*   dst  = (const int*)d_in[3];
  const float* pA_w = (const float*)d_in[4];
  const float* pA_b = (const float*)d_in[5];
  const float* pB_w = (const float*)d_in[6];
  const float* pB_b = (const float*)d_in[7];
  const float* mW1  = (const float*)d_in[8];
  const float* mb1  = (const float*)d_in[9];
  const float* mW2  = (const float*)d_in[10];
  const float* mb2  = (const float*)d_in[11];
  const float* BW   = (const float*)d_in[12];
  const float* Bb   = (const float*)d_in[13];
  const float* CW   = (const float*)d_in[14];
  const float* Cb   = (const float*)d_in[15];

  int N  = in_sizes[0] / DD;
  int NE = in_sizes[2];

  // workspace layout (all 256B aligned): f16 weights | Vp | V1sum | V2sum | deg
  char* ws = (char*)d_ws;
  _Float16* w16 = (_Float16*)ws;                    // 6*4096*2 = 49152 B
  float* Vp  = (float*)(ws + 49152);
  float* V1s = Vp + (size_t)N * DD;
  float* V2s = V1s + (size_t)N * DD;
  float* deg = V2s + (size_t)N * DD;

  convert_w_kernel<<<16, 256, 0, stream>>>(pA_w, pB_w, mW1, mW2, BW, CW, w16);
  zero_kernel<<<1024, 256, 0, stream>>>(V1s, (size_t)N * DD * 2 + (size_t)N);

  int ntiles = (N + 15) / 16;
  node_pool_kernel<<<ntiles, 32, 0, stream>>>(V, w16, pA_b, pB_b, Vp, N);

  degree_kernel<<<(NE + 255) / 256, 256, 0, stream>>>(dst, deg, NE);

  int etiles = (NE + 15) / 16;
  edge_msg_kernel<<<etiles, 32, 0, stream>>>(E, src, dst, w16, mb1, mb2, Bb, Cb,
                                             Vp, V1s, V2s, NE);

  finalize_kernel<<<(N * DD + 255) / 256, 256, 0, stream>>>(V1s, V2s, deg,
                                                            (float*)d_out, N);
}
